// cINN_39419209843060
// MI455X (gfx1250) — matrix operand order repaired and checked by reading, not compiled, not run on hardware
//
#include <hip/hip_runtime.h>
#include <cstdint>
#include <cstddef>

// Problem constants (from reference)
#define D_     784
#define HALF_  392
#define HID_   512
#define COND_  10
#define NBLK_  20
#define BATCH  4096
#define KP1    416   // 392+10=402 padded up to multiple of 32

typedef __attribute__((ext_vector_type(16))) _Float16 v16h;
typedef __attribute__((ext_vector_type(8)))  _Float16 v8h;
typedef __attribute__((ext_vector_type(8)))  float    v8f;

#define BM 128
#define BN 128
#define BK 32

// ---------------------------------------------------------------------------
// Weight convert: f32 (nblk, Kin, N) -> f16 transposed+padded (nblk, N, Kp)
// rows k >= Kin are zero.
// ---------------------------------------------------------------------------
__global__ void wconv_kernel(const float* __restrict__ W, _Float16* __restrict__ Wt,
                             int Kin, int Kp, int N)
{
    size_t idx   = (size_t)blockIdx.x * blockDim.x + threadIdx.x;
    size_t total = (size_t)NBLK_ * N * Kp;
    if (idx >= total) return;
    int k   = (int)(idx % Kp);
    size_t t = idx / Kp;
    int n   = (int)(t % N);
    int blk = (int)(t / N);
    float v = (k < Kin) ? W[((size_t)blk * Kin + k) * N + n] : 0.0f;
    Wt[idx] = (_Float16)v;
}

// ---------------------------------------------------------------------------
// Init: zero jac, write constant one-hot/zero tails (cols 392..415) of both
// Xcat staging buffers (the condition vector never changes across blocks).
// ---------------------------------------------------------------------------
__global__ void init_kernel(const int* __restrict__ l,
                            _Float16* __restrict__ xc1, _Float16* __restrict__ xc2,
                            float* __restrict__ jac)
{
    int b = blockIdx.x * blockDim.x + threadIdx.x;
    if (b >= BATCH) return;
    jac[b] = 0.0f;
    int lab = l[b];
    for (int k = HALF_; k < KP1; ++k) {
        _Float16 v = (_Float16)((k - HALF_ == lab) ? 1.0f : 0.0f);
        xc1[(size_t)b * KP1 + k] = v;
        xc2[(size_t)b * KP1 + k] = v;
    }
}

// ---------------------------------------------------------------------------
// Permute: vperm = vsrc[:, perm]; x1 = vperm[:, :392], x2 = vperm[:, 392:],
// and seed Xcat2 (f16) with x2.
// ---------------------------------------------------------------------------
__global__ void permute_kernel(const float* __restrict__ vsrc, const int* __restrict__ perm,
                               float* __restrict__ x1, float* __restrict__ x2,
                               _Float16* __restrict__ xcat2)
{
    int b = blockIdx.x;
    for (int j = threadIdx.x; j < D_; j += blockDim.x) {
        float val = vsrc[(size_t)b * D_ + perm[j]];
        if (j < HALF_) {
            x1[(size_t)b * HALF_ + j] = val;
        } else {
            int jj = j - HALF_;
            x2[(size_t)b * HALF_ + jj] = val;
            xcat2[(size_t)b * KP1 + jj] = (_Float16)val;
        }
    }
}

// ---------------------------------------------------------------------------
// WMMA GEMM: out = A(MxKp, f16, ld=Kp) @ Wt^T + bias
//   Wt is stored transposed: (N_pad rows) x Kp, f16 (rows >= N are zero-guarded)
// mode 0: Hout = relu(out) as f16 (ld = N)
// mode 1: Rout = out as f32 (ld = N)
// 256 threads = 8 waves, wave grid 4(M) x 2(N); each wave: 32x64 tile
// ---------------------------------------------------------------------------
__global__ __launch_bounds__(256) void gemm_wmma_kernel(
    const _Float16* __restrict__ A, int Kp,
    const _Float16* __restrict__ Wt,
    const float* __restrict__ bias,
    int N,
    _Float16* __restrict__ Hout,
    float* __restrict__ Rout,
    int mode)
{
    __shared__ __align__(16) _Float16 As[BM * BK];
    __shared__ __align__(16) _Float16 Bs[BN * BK];

    const int tid  = threadIdx.x;
    const int wave = tid >> 5;
    const int lane = tid & 31;
    const int lo   = lane & 15;
    const int hi   = lane >> 4;
    const int wm   = wave & 3;   // 0..3 -> M
    const int wn   = wave >> 2;  // 0..1 -> N

    const int m0 = blockIdx.y * BM;
    const int n0 = blockIdx.x * BN;

    const v8f vzero = {0.f, 0.f, 0.f, 0.f, 0.f, 0.f, 0.f, 0.f};
    v8f acc[2][4];
    for (int i = 0; i < 2; ++i)
        for (int j = 0; j < 4; ++j)
            acc[i][j] = vzero;

    // tile-fill assignment: 2 threads per row, 16 halves (32B) each
    const int ar = tid & 127;
    const int ac = (tid >> 7) << 4;  // 0 or 16

    for (int k0 = 0; k0 < Kp; k0 += BK) {
        __syncthreads();
        // A tile: rows m0+ar always < M (M multiple of BM)
        {
            const uint4* gp = (const uint4*)(A + (size_t)(m0 + ar) * Kp + k0 + ac);
            uint4 q0 = gp[0];
            uint4 q1 = gp[1];
            *(uint4*)(&As[ar * BK + ac])     = q0;
            *(uint4*)(&As[ar * BK + ac + 8]) = q1;
        }
        // B tile (Wt rows are output columns n); zero-pad rows beyond N
        {
            uint4 q0 = {0, 0, 0, 0}, q1 = {0, 0, 0, 0};
            int n = n0 + ar;
            if (n < N) {
                const uint4* gp = (const uint4*)(Wt + (size_t)n * Kp + k0 + ac);
                q0 = gp[0];
                q1 = gp[1];
            }
            *(uint4*)(&Bs[ar * BK + ac])     = q0;
            *(uint4*)(&Bs[ar * BK + ac + 8]) = q1;
        }
        __syncthreads();

        // Fragments.
        // A 16x32 f16 layout: half e <-> K = ((e>>3)<<4) + hi*8 + (e&7)
        //   -> contiguous 8 halves at k=hi*8 and k=16+hi*8
        v16h afr[2];
#pragma unroll
        for (int mf = 0; mf < 2; ++mf) {
            int row = (wm << 5) + (mf << 4) + lo;
            v8h p0 = *(const v8h*)(&As[row * BK + hi * 8]);
            v8h p1 = *(const v8h*)(&As[row * BK + 16 + hi * 8]);
            afr[mf] = __builtin_shufflevector(p0, p1, 0, 1, 2, 3, 4, 5, 6, 7,
                                              8, 9, 10, 11, 12, 13, 14, 15);
        }
        // B 32x16 f16 layout: half e <-> K = ((e>>3)<<4) + hi*8 + (e&7), N = lo
        //   -> contiguous 8 halves at k=hi*8 and k=16+hi*8
        v16h bfr[4];
#pragma unroll
        for (int nf = 0; nf < 4; ++nf) {
            int row = (wn << 6) + (nf << 4) + lo;
            v8h p0 = *(const v8h*)(&Bs[row * BK + hi * 8]);
            v8h p1 = *(const v8h*)(&Bs[row * BK + 16 + hi * 8]);
            bfr[nf] = __builtin_shufflevector(p0, p1, 0, 1, 2, 3, 4, 5, 6, 7,
                                              8, 9, 10, 11, 12, 13, 14, 15);
        }
#pragma unroll
        for (int mf = 0; mf < 2; ++mf)
#pragma unroll
            for (int nf = 0; nf < 4; ++nf)
                acc[mf][nf] = __builtin_amdgcn_wmma_f32_16x16x32_f16(
                    false, afr[mf], false, bfr[nf], (short)0, acc[mf][nf], false, false);
    }

    // Epilogue. D layout: element r <-> M = r + hi*8, N = lo.
#pragma unroll
    for (int mf = 0; mf < 2; ++mf) {
#pragma unroll
        for (int nf = 0; nf < 4; ++nf) {
            int n = n0 + (wn << 6) + (nf << 4) + lo;
            if (n >= N) continue;
            float bv    = bias[n];
            int   mbase = m0 + (wm << 5) + (mf << 4) + hi * 8;
            if (mode == 0) {
#pragma unroll
                for (int r = 0; r < 8; ++r) {
                    float val = acc[mf][nf][r] + bv;
                    Hout[(size_t)(mbase + r) * N + n] = (_Float16)fmaxf(val, 0.0f);
                }
            } else {
#pragma unroll
                for (int r = 0; r < 8; ++r) {
                    float val = acc[mf][nf][r] + bv;
                    Rout[(size_t)(mbase + r) * N + n] = val;
                }
            }
        }
    }
}

// ---------------------------------------------------------------------------
// Coupling: s = R[:, :392], t = R[:, 392:]; ls = 0.636*atan(s);
// y = exp(ls)*xpass + t  -> vdst[:, dstOff..dstOff+392)
// optionally seed next Xcat with f16(y); jac[b] += sum_j ls
// one block (256 thr) per batch row
// ---------------------------------------------------------------------------
__global__ void coupling_kernel(const float* __restrict__ R,
                                const float* __restrict__ xpass,
                                float* __restrict__ vdst,
                                _Float16* __restrict__ xcatNext,
                                float* __restrict__ jac,
                                int dstOff)
{
    int b = blockIdx.x;
    __shared__ float red[256];
    float lsum = 0.0f;
    for (int j = threadIdx.x; j < HALF_; j += 256) {
        float s  = R[(size_t)b * D_ + j];
        float t  = R[(size_t)b * D_ + HALF_ + j];
        float ls = 0.636f * atanf(s);
        float y  = expf(ls) * xpass[(size_t)b * HALF_ + j] + t;
        vdst[(size_t)b * D_ + dstOff + j] = y;
        if (xcatNext) xcatNext[(size_t)b * KP1 + j] = (_Float16)y;
        lsum += ls;
    }
    red[threadIdx.x] = lsum;
    __syncthreads();
    for (int s = 128; s > 0; s >>= 1) {
        if (threadIdx.x < s) red[threadIdx.x] += red[threadIdx.x + s];
        __syncthreads();
    }
    if (threadIdx.x == 0) jac[b] += red[0];
}

// ---------------------------------------------------------------------------
// Finalize: d_out = [z (B*784), jac (B)]
// ---------------------------------------------------------------------------
__global__ void finalize_kernel(const float* __restrict__ v, const float* __restrict__ jac,
                                float* __restrict__ out)
{
    size_t i = (size_t)blockIdx.x * blockDim.x + threadIdx.x;
    const size_t nz = (size_t)BATCH * D_;
    if (i < nz) out[i] = v[i];
    else if (i < nz + BATCH) out[i] = jac[i - nz];
}

// ---------------------------------------------------------------------------
extern "C" void kernel_launch(void* const* d_in, const int* in_sizes, int n_in,
                              void* d_out, int out_size, void* d_ws, size_t ws_size,
                              hipStream_t stream)
{
    const float* x    = (const float*)d_in[0];
    const int*   l    = (const int*)d_in[1];
    const int*   perms= (const int*)d_in[2];
    const float* s1W1 = (const float*)d_in[3];
    const float* s1b1 = (const float*)d_in[4];
    const float* s1W2 = (const float*)d_in[5];
    const float* s1b2 = (const float*)d_in[6];
    const float* s2W1 = (const float*)d_in[7];
    const float* s2b1 = (const float*)d_in[8];
    const float* s2W2 = (const float*)d_in[9];
    const float* s2b2 = (const float*)d_in[10];

    size_t off = 0;
    auto alloc = [&](size_t bytes) -> void* {
        off = (off + 255) & ~(size_t)255;
        void* p = (char*)d_ws + off;
        off += bytes;
        return p;
    };

    const size_t w1e = (size_t)NBLK_ * HID_ * KP1;   // per-tensor f16 elems, layer1 (transposed+padded)
    const size_t w2e = (size_t)NBLK_ * D_ * HID_;    // layer2 (transposed)

    _Float16* Wt_s1W1 = (_Float16*)alloc(w1e * 2);
    _Float16* Wt_s2W1 = (_Float16*)alloc(w1e * 2);
    _Float16* Wt_s1W2 = (_Float16*)alloc(w2e * 2);
    _Float16* Wt_s2W2 = (_Float16*)alloc(w2e * 2);
    _Float16* xcat1   = (_Float16*)alloc((size_t)BATCH * KP1 * 2);
    _Float16* xcat2   = (_Float16*)alloc((size_t)BATCH * KP1 * 2);
    _Float16* Hbuf    = (_Float16*)alloc((size_t)BATCH * HID_ * 2);
    float*    Rbuf    = (float*)   alloc((size_t)BATCH * D_ * 4);
    float*    x1buf   = (float*)   alloc((size_t)BATCH * HALF_ * 4);
    float*    x2buf   = (float*)   alloc((size_t)BATCH * HALF_ * 4);
    float*    v0      = (float*)   alloc((size_t)BATCH * D_ * 4);
    float*    v1      = (float*)   alloc((size_t)BATCH * D_ * 4);
    float*    jac     = (float*)   alloc((size_t)BATCH * 4);

    // 1) weight convert (transpose + f16 + K-pad)
    {
        int thr = 256;
        int g1  = (int)((w1e + thr - 1) / thr);
        int g2  = (int)((w2e + thr - 1) / thr);
        wconv_kernel<<<g1, thr, 0, stream>>>(s1W1, Wt_s1W1, HALF_ + COND_, KP1, HID_);
        wconv_kernel<<<g1, thr, 0, stream>>>(s2W1, Wt_s2W1, HALF_ + COND_, KP1, HID_);
        wconv_kernel<<<g2, thr, 0, stream>>>(s1W2, Wt_s1W2, HID_, HID_, D_);
        wconv_kernel<<<g2, thr, 0, stream>>>(s2W2, Wt_s2W2, HID_, HID_, D_);
    }
    // 2) init (jac=0, constant one-hot tails of Xcat buffers)
    init_kernel<<<BATCH / 256, 256, 0, stream>>>(l, xcat1, xcat2, jac);

    dim3 gL1((HID_ + BN - 1) / BN, BATCH / BM);  // 4 x 32
    dim3 gL2((D_ + BN - 1) / BN, BATCH / BM);    // 7 x 32

    float* vbufs[2] = {v0, v1};
    const float* vsrc = x;
    for (int bi = 0; bi < NBLK_; ++bi) {
        float* vdst = vbufs[bi & 1];

        permute_kernel<<<BATCH, 256, 0, stream>>>(vsrc, perms + (size_t)bi * D_,
                                                  x1buf, x2buf, xcat2);
        // subnet "2": r2 = relu([x2,c] @ s2W1 + s2b1) @ s2W2 + s2b2
        gemm_wmma_kernel<<<gL1, 256, 0, stream>>>(xcat2, KP1,
                                                  Wt_s2W1 + (size_t)bi * HID_ * KP1,
                                                  s2b1 + (size_t)bi * HID_, HID_,
                                                  Hbuf, nullptr, 0);
        gemm_wmma_kernel<<<gL2, 256, 0, stream>>>(Hbuf, HID_,
                                                  Wt_s2W2 + (size_t)bi * D_ * HID_,
                                                  s2b2 + (size_t)bi * D_, D_,
                                                  nullptr, Rbuf, 1);
        // y1 = exp(ls2)*x1 + t2 ; seeds xcat1
        coupling_kernel<<<BATCH, 256, 0, stream>>>(Rbuf, x1buf, vdst, xcat1, jac, 0);

        // subnet "1": r1 = relu([y1,c] @ s1W1 + s1b1) @ s1W2 + s1b2
        gemm_wmma_kernel<<<gL1, 256, 0, stream>>>(xcat1, KP1,
                                                  Wt_s1W1 + (size_t)bi * HID_ * KP1,
                                                  s1b1 + (size_t)bi * HID_, HID_,
                                                  Hbuf, nullptr, 0);
        gemm_wmma_kernel<<<gL2, 256, 0, stream>>>(Hbuf, HID_,
                                                  Wt_s1W2 + (size_t)bi * D_ * HID_,
                                                  s1b2 + (size_t)bi * D_, D_,
                                                  nullptr, Rbuf, 1);
        // y2 = exp(ls1)*x2 + t1
        coupling_kernel<<<BATCH, 256, 0, stream>>>(Rbuf, x2buf, vdst, nullptr, jac, HALF_);

        vsrc = vdst;
    }

    size_t total = (size_t)BATCH * D_ + BATCH;
    finalize_kernel<<<(int)((total + 255) / 256), 256, 0, stream>>>(vsrc, jac, (float*)d_out);
}
